// DCVQQuantizer_17892833755572
// MI455X (gfx1250) — compile-verified, matrix operand order checked by reading, not run on hardware
//
#include <hip/hip_runtime.h>
#include <hip/hip_bf16.h>

typedef __attribute__((ext_vector_type(16))) __bf16 v16bf;
typedef __attribute__((ext_vector_type(8)))  float  v8f;

// Problem constants (from reference): z [16,512,32,32], codebooks [16,1024,32]
#define BATCH   16
#define DCH     512
#define HW      1024          // 32*32
#define NSUB    16
#define MCODES  1024
#define DS      32
#define TTOK    16384         // BATCH*HW
#define ZQ_ELEMS 8388608      // 16*512*32*32
#define VQ_OFF   ZQ_ELEMS
#define IDX_OFF  (ZQ_ELEMS + 1)
#define LOSS_SCALE (1.25f / 8388608.0f)   // (1+beta)/ (N*T*ds)

// ws layout: [0, 1MB)  packed bf16 codebooks in WMMA-B register order
//            [1MB, 1MB+64KB) fp32 ||c||^2 per (n, code)
#define WS_PACK_ELEMS (NSUB * 64 * 32 * 16)   // 524288 bf16 = 1 MB
#define WS_CNORM_OFF  (WS_PACK_ELEMS * 2)     // byte offset

// K index held by (lane, element e) for 16-bit 16x32 A (and mirrored B) WMMA layout.
__device__ __forceinline__ int kmap(int lane, int e) {
    return (lane < 16) ? ((e < 8) ? e : e + 8)
                       : ((e < 8) ? e + 8 : e + 16);
}

// ---------------------------------------------------------------------------
// Prep: pack codebooks (fp32 -> bf16, WMMA B layout), compute ||c||^2, zero vq
// ---------------------------------------------------------------------------
__global__ void dcvq_prep_kernel(const float* __restrict__ codebooks,
                                 __bf16* __restrict__ wsPack,
                                 float* __restrict__ wsCnorm,
                                 float* __restrict__ dOut) {
    int tid = blockIdx.x * blockDim.x + threadIdx.x;   // 0 .. 32767

    // ---- pack: one thread per (n, code-tile j, lane) ----
    {
        int n    = tid >> 11;          // 2048 threads per subspace
        int rem  = tid & 2047;
        int j    = rem >> 5;           // code tile 0..63
        int lane = rem & 31;
        int code = j * 16 + (lane & 15);
        const float* src = codebooks + ((size_t)n * MCODES + code) * DS;
        v16bf pk;
#pragma unroll
        for (int e = 0; e < 16; ++e) {
            pk[e] = (__bf16)src[kmap(lane, e)];
        }
        *(v16bf*)(wsPack + (size_t)tid * 16) = pk;
    }

    // ---- cnorm: one thread per (n, code) for tid < 16384 ----
    if (tid < NSUB * MCODES) {
        const float* src = codebooks + (size_t)tid * DS;
        float s = 0.f;
#pragma unroll
        for (int k = 0; k < DS; ++k) { float v = src[k]; s = fmaf(v, v, s); }
        wsCnorm[tid] = s;
    }

    if (tid == 0) dOut[VQ_OFF] = 0.0f;
}

// ---------------------------------------------------------------------------
// Main: per (128-token tile, subspace): WMMA distances, argmin, gather, loss
// ---------------------------------------------------------------------------
__global__ void dcvq_main_kernel(const float* __restrict__ z,
                                 const float* __restrict__ codebooks,
                                 const __bf16* __restrict__ wsPack,
                                 const float* __restrict__ wsCnorm,
                                 float* __restrict__ dOut) {
    const int n    = blockIdx.y;
    const int wave = threadIdx.x >> 5;
    const int lane = threadIdx.x & 31;
    const int laneM = lane & 15;

    const int tokBase = blockIdx.x * 128 + wave * 16;   // 16 tokens per wave
    const int b   = tokBase >> 10;                      // image index
    const int thw = tokBase & 1023;                     // position inside image

    // Base offset of (b, n*32 + 0, thw + laneM) in z / z_q ([B,D,H,W] layout)
    const size_t zTokBase = ((size_t)b * DCH + n * DS) * HW + thw + laneM;

    // ---- load A tile: token row = laneM, 16 k-values per lane (fp32 kept) ----
    float aF[16];
    v16bf a;
#pragma unroll
    for (int e = 0; e < 16; ++e) {
        int k = kmap(lane, e);
        float v = z[zTokBase + (size_t)k * HW];
        aF[e] = v;
        a[e]  = (__bf16)v;
    }

    const v16bf* Bpk = (const v16bf*)wsPack + (size_t)n * 64 * 32;
    const float* cn  = wsCnorm + n * MCODES;

    float bestVal[8];
    int   bestIdx[8];
#pragma unroll
    for (int r = 0; r < 8; ++r) { bestVal[r] = 3.4e38f; bestIdx[r] = 0; }

    const v8f czero = {0.f, 0.f, 0.f, 0.f, 0.f, 0.f, 0.f, 0.f};

    // ---- 64 code tiles: one bf16 WMMA each (K = ds = 32 exactly) ----
#pragma unroll 4
    for (int j = 0; j < 64; ++j) {
        v16bf bm = Bpk[(size_t)j * 32 + lane];
        float cnorm = cn[j * 16 + laneM];
        v8f acc = __builtin_amdgcn_wmma_f32_16x16x32_bf16(
            false, a, false, bm, (short)0, czero, false, false);
        int cand = j * 16 + laneM;
#pragma unroll
        for (int r = 0; r < 8; ++r) {
            float sc = fmaf(-2.0f, acc[r], cnorm);   // ||c||^2 - 2 s.c
            bool lt = sc < bestVal[r];
            bestVal[r] = lt ? sc : bestVal[r];
            bestIdx[r] = lt ? cand : bestIdx[r];
        }
    }

    // ---- argmin reduce across the 16 lanes sharing each token row ----
#pragma unroll
    for (int off = 1; off < 16; off <<= 1) {
#pragma unroll
        for (int r = 0; r < 8; ++r) {
            float ov = __shfl_xor(bestVal[r], off, 32);
            int   oi = __shfl_xor(bestIdx[r], off, 32);
            bool take = (ov < bestVal[r]) || (ov == bestVal[r] && oi < bestIdx[r]);
            bestVal[r] = take ? ov : bestVal[r];
            bestIdx[r] = take ? oi : bestIdx[r];
        }
    }
    // lanes 0-15 now hold argmin for rows M=0..7 (slot r), lanes 16-31 for M=8..15

    // ---- broadcast winner index for token (lane&15) to every lane; write indices
    int idxLane = 0;
#pragma unroll
    for (int m = 0; m < 16; ++m) {
        int im = __shfl(bestIdx[m & 7], (m & 8) ? 16 : 0, 32);
        if (laneM == m && lane < 16) idxLane = im;
        if (lane == m) {
            dOut[IDX_OFF + ((size_t)b * NSUB + n) * HW + thw + m] = (float)im;
        }
    }
    idxLane = __shfl(idxLane, laneM, 32);   // lanes 16-31 copy from partner lane

    // ---- gather q (fp32), write z_q, accumulate (s - q)^2 ----
    const float* qptr = codebooks + ((size_t)n * MCODES + idxLane) * DS;
    float loss = 0.f;
#pragma unroll
    for (int e = 0; e < 16; ++e) {
        int k = kmap(lane, e);
        float qv = qptr[k];
        float d  = aF[e] - qv;
        loss = fmaf(d, d, loss);
        dOut[zTokBase + (size_t)k * HW] = qv;    // z_q (forward value of STE)
    }

    // ---- wave-reduce loss, single scaled atomic add into vq scalar ----
#pragma unroll
    for (int off = 16; off >= 1; off >>= 1) loss += __shfl_xor(loss, off, 32);
    if (lane == 0) atomicAdd(dOut + VQ_OFF, loss * LOSS_SCALE);
}

// ---------------------------------------------------------------------------
extern "C" void kernel_launch(void* const* d_in, const int* in_sizes, int n_in,
                              void* d_out, int out_size, void* d_ws, size_t ws_size,
                              hipStream_t stream) {
    const float* z         = (const float*)d_in[0];
    const float* codebooks = (const float*)d_in[1];
    float* out = (float*)d_out;

    __bf16* wsPack  = (__bf16*)d_ws;
    float*  wsCnorm = (float*)((char*)d_ws + WS_CNORM_OFF);

    // Prep: 32768 threads (pack) + cnorm + vq zero
    dcvq_prep_kernel<<<dim3(128), dim3(256), 0, stream>>>(codebooks, wsPack, wsCnorm, out);

    // Main: 128 token-tiles x 16 subspaces, 8 waves (16 tokens each) per block
    dcvq_main_kernel<<<dim3(128, 16), dim3(256), 0, stream>>>(z, codebooks, wsPack, wsCnorm, out);
}
